// CausalSelfAttention_79293686219029
// MI455X (gfx1250) — matrix operand order repaired and checked by reading, not compiled, not run on hardware
//
#include <hip/hip_runtime.h>

// ---------------------------------------------------------------------------
// Causal self-attention forward for MI455X (gfx1250, wave32, WMMA).
// B=2, T=2048, D=1024, H=16, HD=64.  fp32 in/out, bf16 WMMA compute,
// fp32 accumulation.  Pipeline:
//   cvt(x,W*) -> gemm<0>(q) gemm<0>(k) gemm<1>(v, stored [b,h,hd,t])
//   -> flash-attention (wave-per-16-row Q tile) -> gemm<2>(out proj, fp32)
// GEMMs stage tiles into LDS with global_load_async_to_lds_b128 (ASYNCcnt),
// double-buffered, and feed WMMA fragments from LDS.
// ---------------------------------------------------------------------------

constexpr int B_  = 2;
constexpr int T_  = 2048;
constexpr int D_  = 1024;
constexpr int H_  = 16;
constexpr int HD_ = 64;
constexpr int M_  = B_ * T_;   // 4096 rows for all GEMMs

constexpr int LP_ = 40;        // LDS row pitch (32 data + 8 pad bf16)

typedef __attribute__((ext_vector_type(16))) __bf16 v16bf;
typedef __attribute__((ext_vector_type(8)))  float  v8f;

union Frag {                 // one 16-bit WMMA operand fragment per lane
  v16bf v;
  uint4 q[2];
};
union Pack8 {                // 8 bf16 -> one 16-byte store
  __bf16 e[8];
  uint4  q;
};

__device__ __forceinline__ v8f wmma_bf16(v16bf a, v16bf b, v8f c) {
  // (neg_a, A, neg_b, B, c_mod, C, reuse_a, reuse_b)
  return __builtin_amdgcn_wmma_f32_16x16x32_bf16(false, a, false, b,
                                                 (short)0, c, false, false);
}

// A-matrix fragment (16x32, 16-bit): lane = row, K chunks at 8*hi and 16+8*hi
__device__ __forceinline__ v16bf frag_a_from(const __bf16* rowk0, int hi) {
  Frag f;
  f.q[0] = *(const uint4*)(rowk0 + 8 * hi);
  f.q[1] = *(const uint4*)(rowk0 + 16 + 8 * hi);
  return f.v;
}
// B-matrix fragment (32x16, 16-bit): lane = column, K chunks at 8*hi and 16+8*hi
__device__ __forceinline__ v16bf frag_b_from(const __bf16* rowk0, int hi) {
  Frag f;
  f.q[0] = *(const uint4*)(rowk0 + 8 * hi);
  f.q[1] = *(const uint4*)(rowk0 + 16 + 8 * hi);
  return f.v;
}

__device__ __forceinline__ v8f vzero8() {
  v8f z;
#pragma unroll
  for (int j = 0; j < 8; ++j) z[j] = 0.0f;
  return z;
}

// LDS byte offset of a generic pointer to __shared__ memory: on gfx1250 the
// flat aperture form is {SHARED_BASE_HI32, lds_offset32} -> low 32 bits.
__device__ __forceinline__ unsigned lds_off(const void* p) {
  return (unsigned)(unsigned long long)p;
}

// Async copy of 16 bytes global -> LDS (GLOBAL_LOAD_ASYNC_TO_LDS_B128,
// GV addressing, tracked by ASYNCcnt).
__device__ __forceinline__ void async_b128(unsigned lds_addr,
                                           const void* gaddr) {
  asm volatile("global_load_async_to_lds_b128 %0, %1, off"
               :
               : "v"(lds_addr), "v"(gaddr)
               : "memory");
}
__device__ __forceinline__ void wait_async0() {
  asm volatile("s_wait_asynccnt 0x0" ::: "memory");
}

// reductions across the 16 lanes of a half-wave (wave32)
__device__ __forceinline__ float hmax16(float v) {
  v = fmaxf(v, __shfl_xor(v, 1, 32));
  v = fmaxf(v, __shfl_xor(v, 2, 32));
  v = fmaxf(v, __shfl_xor(v, 4, 32));
  v = fmaxf(v, __shfl_xor(v, 8, 32));
  return v;
}
__device__ __forceinline__ float hsum16(float v) {
  v += __shfl_xor(v, 1, 32);
  v += __shfl_xor(v, 2, 32);
  v += __shfl_xor(v, 4, 32);
  v += __shfl_xor(v, 8, 32);
  return v;
}

// ---------------------------------------------------------------------------
__global__ void cvt_f32_bf16(const float* __restrict__ s, __bf16* __restrict__ d,
                             int n) {
  int i = blockIdx.x * blockDim.x + threadIdx.x;
  int stride = gridDim.x * blockDim.x;
  for (; i < n; i += stride) d[i] = (__bf16)s[i];
}

// ---------------------------------------------------------------------------
// C[m,n] = sum_k A[m,k] * W[n,k] + bias[n]       (A: M_ x D_, W: D_ x D_)
// MODE 0: store bf16 row-major [M_, D_]
// MODE 1: store bf16 transposed per head [b, h, hd, t]   (for V)
// MODE 2: store fp32 row-major [M_, D_]                  (final output)
// Block = 4 waves, block tile 128x128, wave tile 64x64 = 4x4 WMMA accums.
// K-loop step 32: async-stage A/W 128x32 tiles to LDS, double buffered.
// ---------------------------------------------------------------------------
template <int MODE>
__global__ __launch_bounds__(128) void gemm_kernel(
    const __bf16* __restrict__ A, const __bf16* __restrict__ W,
    const float* __restrict__ bias, void* __restrict__ out) {
  __shared__ __align__(16) __bf16 bufA[2][128 * LP_];
  __shared__ __align__(16) __bf16 bufW[2][128 * LP_];

  const int tid  = threadIdx.x;
  const int lane = tid & 31;
  const int wid  = tid >> 5;
  const int hi   = lane >> 4;
  const int ln   = lane & 15;
  const int m_blk = blockIdx.y * 128;
  const int n_blk = blockIdx.x * 128;
  const int mw = (wid >> 1) * 64;   // wave tile origin inside block tile
  const int nw = (wid & 1) * 64;

  // stage one 128x32 tile of A and of W into LDS buffer `buf` (async)
  auto stage = [&](int buf, int k0) {
#pragma unroll
    for (int c = 0; c < 4; ++c) {
      async_b128(lds_off(&bufA[buf][tid * LP_ + c * 8]),
                 A + (size_t)(m_blk + tid) * D_ + k0 + c * 8);
      async_b128(lds_off(&bufW[buf][tid * LP_ + c * 8]),
                 W + (size_t)(n_blk + tid) * D_ + k0 + c * 8);
    }
  };

  v8f acc[4][4];
#pragma unroll
  for (int i = 0; i < 4; ++i)
#pragma unroll
    for (int j = 0; j < 4; ++j) acc[i][j] = vzero8();

  stage(0, 0);
  for (int k0 = 0; k0 < D_; k0 += 32) {
    const int cur = (k0 >> 5) & 1;
    wait_async0();        // prefetched tile for `cur` landed in LDS
    __syncthreads();      // everyone done reading the other buffer
    if (k0 + 32 < D_) stage(cur ^ 1, k0 + 32);

    v16bf af[4], wf[4];
#pragma unroll
    for (int i = 0; i < 4; ++i)
      af[i] = frag_a_from(&bufA[cur][(mw + i * 16 + ln) * LP_], hi);
#pragma unroll
    for (int j = 0; j < 4; ++j)
      wf[j] = frag_b_from(&bufW[cur][(nw + j * 16 + ln) * LP_], hi);

#pragma unroll
    for (int i = 0; i < 4; ++i)
#pragma unroll
      for (int j = 0; j < 4; ++j)
        acc[i][j] = wmma_bf16(af[i], wf[j], acc[i][j]);
  }

#pragma unroll
  for (int i = 0; i < 4; ++i) {
#pragma unroll
    for (int jf = 0; jf < 4; ++jf) {
      const int   n  = n_blk + nw + jf * 16 + ln;
      const float bn = bias[n];
      if constexpr (MODE == 0) {
        __bf16* Ob = (__bf16*)out;
#pragma unroll
        for (int j = 0; j < 8; ++j) {
          const int m = m_blk + mw + i * 16 + 8 * hi + j;
          Ob[(size_t)m * D_ + n] = (__bf16)(acc[i][jf][j] + bn);
        }
      } else if constexpr (MODE == 1) {
        __bf16* Ov = (__bf16*)out;
        const int mt = m_blk + mw + i * 16;
        const int bb = mt >> 11;              // / T_
        const int t0 = (mt & (T_ - 1)) + 8 * hi;
        const int hh = n >> 6;                // / HD_
        const int hd = n & (HD_ - 1);
        Pack8 p;
#pragma unroll
        for (int j = 0; j < 8; ++j) p.e[j] = (__bf16)(acc[i][jf][j] + bn);
        *(uint4*)&Ov[((size_t)(bb * H_ + hh) * HD_ + hd) * T_ + t0] = p.q;
      } else {
        float* Of = (float*)out;
#pragma unroll
        for (int j = 0; j < 8; ++j) {
          const int m = m_blk + mw + i * 16 + 8 * hi + j;
          Of[(size_t)m * D_ + n] = acc[i][jf][j] + bn;
        }
      }
    }
  }
}

// ---------------------------------------------------------------------------
// Flash attention: one wave handles a 16-row Q tile of one (b,h).
// KV processed in 32-column blocks; S via 4 WMMAs, P via LDS transpose
// into A-layout, P@V via 4 WMMAs against transposed V [b,h,hd,t].
// ---------------------------------------------------------------------------
__global__ __launch_bounds__(128) void attn_kernel(
    const __bf16* __restrict__ qm, const __bf16* __restrict__ km,
    const __bf16* __restrict__ vt, __bf16* __restrict__ aout) {
  __shared__ __align__(16) __bf16 ldsP[4][16][LP_];

  const int lane = threadIdx.x & 31;
  const int wid  = threadIdx.x >> 5;
  const int hi   = lane >> 4;
  const int ln   = lane & 15;

  const int wav = blockIdx.x * 4 + wid;
  const int qt  = wav & 127;        // q tile within (b,h)
  const int bh  = wav >> 7;
  const int b   = bh >> 4;
  const int h   = bh & 15;
  const int q0  = qt * 16;

  const __bf16* qbase = qm + (size_t)b * T_ * D_ + (size_t)h * HD_;
  const __bf16* kbase = km + (size_t)b * T_ * D_ + (size_t)h * HD_;
  const __bf16* vbase = vt + (size_t)(b * H_ + h) * HD_ * T_;

  const v16bf qa0 = frag_a_from(qbase + (size_t)(q0 + ln) * D_ + 0, hi);
  const v16bf qa1 = frag_a_from(qbase + (size_t)(q0 + ln) * D_ + 32, hi);

  v8f o[4];
#pragma unroll
  for (int f = 0; f < 4; ++f) o[f] = vzero8();
  float mrow[8], lrow[8];
  const float NEG_INF = -__builtin_inff();
#pragma unroll
  for (int j = 0; j < 8; ++j) { mrow[j] = NEG_INF; lrow[j] = 0.0f; }

  __bf16(*P)[LP_] = ldsP[wid];

  for (int s0 = 0; s0 < q0 + 16; s0 += 32) {
    if (s0 + 32 < q0 + 16) {  // pull next KV block toward the WGP
      __builtin_prefetch(kbase + (size_t)(s0 + 32 + ln) * D_, 0, 3);
      __builtin_prefetch(vbase + (size_t)ln * T_ + s0 + 32, 0, 3);
    }

    // ---- S = (Q K^T) * scale, 16x32 tile as two 16x16 f32 tiles ----
    const v16bf kb00 = frag_b_from(kbase + (size_t)(s0 + ln) * D_ + 0, hi);
    const v16bf kb01 = frag_b_from(kbase + (size_t)(s0 + ln) * D_ + 32, hi);
    const v16bf kb10 = frag_b_from(kbase + (size_t)(s0 + 16 + ln) * D_ + 0, hi);
    const v16bf kb11 = frag_b_from(kbase + (size_t)(s0 + 16 + ln) * D_ + 32, hi);

    v8f sc0 = wmma_bf16(qa0, kb00, vzero8());
    sc0     = wmma_bf16(qa1, kb01, sc0);
    v8f sc1 = wmma_bf16(qa0, kb10, vzero8());
    sc1     = wmma_bf16(qa1, kb11, sc1);

#pragma unroll
    for (int j = 0; j < 8; ++j) { sc0[j] *= 0.125f; sc1[j] *= 0.125f; }

    if (s0 + 32 > q0) {  // diagonal block(s): causal mask
#pragma unroll
      for (int j = 0; j < 8; ++j) {
        const int trow = q0 + j + 8 * hi;
        if (s0 + ln > trow)      sc0[j] = NEG_INF;
        if (s0 + 16 + ln > trow) sc1[j] = NEG_INF;
      }
    }

    // ---- online softmax: per-row (j, half-wave) stats ----
    float alph[8];
#pragma unroll
    for (int j = 0; j < 8; ++j) {
      float mc   = hmax16(fmaxf(sc0[j], sc1[j]));
      float mnew = fmaxf(mrow[j], mc);
      float p0   = __expf(sc0[j] - mnew);
      float p1   = __expf(sc1[j] - mnew);
      float rs   = hsum16(p0 + p1);
      float a    = __expf(mrow[j] - mnew);
      lrow[j]    = lrow[j] * a + rs;
      mrow[j]    = mnew;
      alph[j]    = a;
      const int r = j + 8 * hi;
      P[r][ln]      = (__bf16)p0;   // C-layout -> LDS 16x32 bf16 tile
      P[r][16 + ln] = (__bf16)p1;
    }

#pragma unroll
    for (int f = 0; f < 4; ++f)
#pragma unroll
      for (int j = 0; j < 8; ++j) o[f][j] *= alph[j];

    asm volatile("s_wait_dscnt 0" ::: "memory");

    // reload P as an A-matrix fragment (row = ln, chunks 8*hi / 16+8*hi)
    Frag pa;
    pa.q[0] = *(const uint4*)&P[ln][8 * hi];
    pa.q[1] = *(const uint4*)&P[ln][16 + 8 * hi];

    // ---- O += P @ V ----
#pragma unroll
    for (int f = 0; f < 4; ++f) {
      const v16bf vf = frag_b_from(vbase + (size_t)(f * 16 + ln) * T_ + s0, hi);
      o[f] = wmma_bf16(pa.v, vf, o[f]);
    }
  }

  float rinv[8];
#pragma unroll
  for (int j = 0; j < 8; ++j) rinv[j] = 1.0f / lrow[j];
#pragma unroll
  for (int f = 0; f < 4; ++f)
#pragma unroll
    for (int j = 0; j < 8; ++j) {
      const int t = q0 + j + 8 * hi;
      aout[(size_t)(b * T_ + t) * D_ + h * HD_ + f * 16 + ln] =
          (__bf16)(o[f][j] * rinv[j]);
    }
}

// ---------------------------------------------------------------------------
extern "C" void kernel_launch(void* const* d_in, const int* in_sizes, int n_in,
                              void* d_out, int out_size, void* d_ws,
                              size_t ws_size, hipStream_t stream) {
  (void)in_sizes; (void)n_in; (void)out_size; (void)ws_size;

  const float* x  = (const float*)d_in[0];
  const float* Wq = (const float*)d_in[1];
  const float* bq = (const float*)d_in[2];
  const float* Wk = (const float*)d_in[3];
  const float* bk = (const float*)d_in[4];
  const float* Wv = (const float*)d_in[5];
  const float* bv = (const float*)d_in[6];
  const float* Wo = (const float*)d_in[7];
  const float* bo = (const float*)d_in[8];

  char* ws = (char*)d_ws;
  const size_t MB = 1024u * 1024u;
  __bf16* xb  = (__bf16*)(ws + 0 * MB);    // 8 MiB  x  bf16
  __bf16* qb  = (__bf16*)(ws + 8 * MB);    // 8 MiB  q  [B*T, D]
  __bf16* kbf = (__bf16*)(ws + 16 * MB);   // 8 MiB  k  [B*T, D]
  __bf16* vtb = (__bf16*)(ws + 24 * MB);   // 8 MiB  v  [B, H, HD, T]
  __bf16* ab  = (__bf16*)(ws + 32 * MB);   // 8 MiB  attn out [B*T, D]
  __bf16* wqb = (__bf16*)(ws + 40 * MB);   // 2 MiB each
  __bf16* wkb = (__bf16*)(ws + 42 * MB);
  __bf16* wvb = (__bf16*)(ws + 44 * MB);
  __bf16* wob = (__bf16*)(ws + 46 * MB);

  cvt_f32_bf16<<<512, 256, 0, stream>>>(x, xb, M_ * D_);
  cvt_f32_bf16<<<256, 256, 0, stream>>>(Wq, wqb, D_ * D_);
  cvt_f32_bf16<<<256, 256, 0, stream>>>(Wk, wkb, D_ * D_);
  cvt_f32_bf16<<<256, 256, 0, stream>>>(Wv, wvb, D_ * D_);
  cvt_f32_bf16<<<256, 256, 0, stream>>>(Wo, wob, D_ * D_);

  const dim3 gg(D_ / 128, M_ / 128);  // (8, 32)
  gemm_kernel<0><<<gg, 128, 0, stream>>>(xb, wqb, bq, qb);
  gemm_kernel<0><<<gg, 128, 0, stream>>>(xb, wkb, bk, kbf);
  gemm_kernel<1><<<gg, 128, 0, stream>>>(xb, wvb, bv, vtb);

  const int nwaves = (M_ / 16) * H_;      // 4096 waves, 4 per block
  attn_kernel<<<nwaves / 4, 128, 0, stream>>>(qb, kbf, vtb, ab);

  gemm_kernel<2><<<gg, 128, 0, stream>>>(ab, wob, bo, d_out);
}
